// Head_10685878632972
// MI455X (gfx1250) — matrix operand order repaired and checked by reading, not compiled, not run on hardware
//
#include <hip/hip_runtime.h>
#include <cstdint>

#define B_ 8
#define T_ 2048
#define C_ 1024
#define H_ 64

typedef __attribute__((ext_vector_type(16))) __bf16 v16bf;
typedef __attribute__((ext_vector_type(8)))  __bf16 v8bf;
typedef __attribute__((ext_vector_type(8)))  float  v8f;
typedef __attribute__((ext_vector_type(4)))  float  v4f;

__device__ __forceinline__ v8f wmma_bf16(v16bf a, v16bf b, v8f c) {
  // D = A(16x32) * B(32x16) + C, f32 accumulate
  return __builtin_amdgcn_wmma_f32_16x16x32_bf16(false, a, false, b, (short)0, c, false, false);
}

// CDNA5 async copy: global -> LDS, 16 bytes per lane, tracked by ASYNCcnt.
__device__ __forceinline__ void async_b128(unsigned lds_off, int goff, uint64_t gbase) {
  asm volatile("global_load_async_to_lds_b128 %0, %1, %2"
               :: "v"(lds_off), "v"(goff), "s"(gbase) : "memory");
}
__device__ __forceinline__ void wait_async0() {
#if __has_builtin(__builtin_amdgcn_s_wait_asynccnt)
  __builtin_amdgcn_s_wait_asynccnt(0);
#else
  asm volatile("s_wait_asynccnt 0x0" ::: "memory");
#endif
}

// ---------------- Kernel 0: weights f32 -> bf16 (fold C^-0.5 into Wq) ----------
__global__ void wconv_kernel(const float* __restrict__ Wq, const float* __restrict__ Wk,
                             const float* __restrict__ Wv, __bf16* __restrict__ Wb) {
  int i = blockIdx.x * blockDim.x + threadIdx.x;
  const int HC = H_ * C_;
  if (i >= HC) return;
  Wb[i]          = (__bf16)(Wq[i] * 0.03125f);   // 1/sqrt(C) = 1/32 exactly
  Wb[HC + i]     = (__bf16)Wk[i];
  Wb[2 * HC + i] = (__bf16)Wv[i];
}

// ---------------- Kernel 1: QKV projection (WMMA bf16) -------------------------
// Each wave: one 16-row M tile, all 3 weights x 4 h-tiles (12 accumulators).
// W k-chunks staged in LDS via double-buffered async copy (issued 1 step ahead);
// x (A operand) pipelined 1 k-step ahead in registers; B ds-loads pipelined one
// WMMA ahead so every s_wait_dscnt covers a load issued a full WMMA earlier.
__global__ __launch_bounds__(128) void qkv_kernel(const float* __restrict__ x,
                                                  const __bf16* __restrict__ Wb,
                                                  __bf16* __restrict__ qb,
                                                  __bf16* __restrict__ kb,
                                                  __bf16* __restrict__ vT) {
  __shared__ __align__(32) __bf16 wsm[2 * 6144];   // 2 bufs x (3*64 rows x 32 dims) = 2 x 12KB
  const int tid  = threadIdx.x;
  const int lane = tid & 31;
  const int wave = tid >> 5;
  const int l16  = lane & 15;
  const int g    = lane >> 4;
  const int mbase = blockIdx.x * 64 + wave * 16;     // 256 blocks * 4 waves
  const unsigned smem_base = (unsigned)(uintptr_t)&wsm[0];  // low 32 bits = LDS offset
  const uint64_t wbase = (uint64_t)(uintptr_t)Wb;

  // Copy the 32-dim k-chunk of all 3*64 W rows into LDS buffer bufi.
  auto issueW = [&](int kk, int bufi) {
    unsigned base = smem_base + bufi * 12288;
#pragma unroll
    for (int i = 0; i < 6; ++i) {
      int f   = i * 128 + tid;            // 0..767 : 16B packets
      int row = f >> 2;                   // 0..191 = widx*64 + h
      int seg = (f & 3) * 16;             // byte seg within 64B row chunk
      async_b128(base + f * 16, row * (C_ * 2) + kk * 2 + seg, wbase);
    }
  };

  v8f acc[12] = {};                       // [widx*4 + nt]
  const float* xrow = x + (size_t)(mbase + l16) * C_;  // A-matrix: lane holds row M=l16

  // prologue: x raw for kk=0, W chunk 0
  v4f r0 = *(const v4f*)(xrow + 16 * g);
  v4f r1 = *(const v4f*)(xrow + 16 * g + 4);
  v4f r2 = *(const v4f*)(xrow + 16 * g + 8);
  v4f r3 = *(const v4f*)(xrow + 16 * g + 12);
  issueW(0, 0);

  for (int kk = 0; kk < C_; kk += 32) {
    wait_async0();
    __syncthreads();

    // Build A (16x32 bf16): e0..15 -> c = kk+16g+e
    v16bf A;
#pragma unroll
    for (int e = 0; e < 4; ++e) {
      A[e]      = (__bf16)r0[e];  A[4 + e]  = (__bf16)r1[e];
      A[8 + e]  = (__bf16)r2[e];  A[12 + e] = (__bf16)r3[e];
    }

    // Pipeline next step: async W chunk + x raw loads issued before the WMMA chain
    v4f n0, n1, n2, n3;
    const bool more = (kk + 32) < C_;
    if (more) {
      issueW(kk + 32, ((kk >> 5) + 1) & 1);
      const float* xp = xrow + kk + 32;
      n0 = *(const v4f*)(xp + 16 * g);
      n1 = *(const v4f*)(xp + 16 * g + 4);
      n2 = *(const v4f*)(xp + 16 * g + 8);
      n3 = *(const v4f*)(xp + 16 * g + 12);
      __builtin_prefetch(xp + 256, 0, 1);          // global_prefetch_b8 further ahead
    }

    // 12-step WMMA chain, B operand ds-load pipelined one step ahead.
    const __bf16* wlds = wsm + ((kk >> 5) & 1) * 6144;
    auto bptr = [&](int s) {   // s = widx*4 + nt ; lane holds col h = 16*nt + l16
      int widx = s >> 2, nt = s & 3;
      return wlds + (widx * 64 + 16 * nt + l16) * 32 + 16 * g;
    };
    v16bf Bcur = *(const v16bf*)bptr(0);
#pragma unroll
    for (int s = 0; s < 12; ++s) {
      v16bf Bnxt;
      if (s < 11) Bnxt = *(const v16bf*)bptr(s + 1);
      acc[s] = wmma_bf16(A, Bcur, acc[s]);
      Bcur = Bnxt;
    }
    if (more) { r0 = n0; r1 = n1; r2 = n2; r3 = n3; }
  }

  // Store: C-layout lane = col h, VGPR r = row (r + 8g)
#pragma unroll
  for (int nt = 0; nt < 4; ++nt) {
    int h = 16 * nt + l16;
#pragma unroll
    for (int r = 0; r < 8; ++r) {
      int row = mbase + r + 8 * g;                 // flat b*T + t
      qb[(size_t)row * H_ + h] = (__bf16)acc[0 + nt][r];
      kb[(size_t)row * H_ + h] = (__bf16)acc[4 + nt][r];
      int bi = row >> 11, t = row & (T_ - 1);
      vT[((size_t)(bi * H_ + h)) * T_ + t] = (__bf16)acc[8 + nt][r];
    }
  }
}

// ---------------- Kernel 2: flash attention (online softmax, WMMA bf16) --------
__global__ __launch_bounds__(128) void flash_kernel(const __bf16* __restrict__ qb,
                                                    const __bf16* __restrict__ kb,
                                                    const __bf16* __restrict__ vT,
                                                    float* __restrict__ out) {
  __shared__ __align__(32) __bf16 smem[16384];  // 2 bufs x (K 64x64 + V^T 64x64) bf16 = 32KB
  const int tid  = threadIdx.x;
  const int lane = tid & 31;
  const int wave = tid >> 5;
  const int l16  = lane & 15;
  const int g    = lane >> 4;
  const int b     = blockIdx.y;
  const int qbase = blockIdx.x * 64;
  const int qrow  = qbase + wave * 16 + l16;     // this lane's q column (S^T layout)
  const unsigned smem_base = (unsigned)(uintptr_t)&smem[0];  // low 32 bits = LDS offset

  // q as B-operand for S^T = K * q^T: lane holds col q; element e -> dim 32c+16g+e
  v16bf qop[2];
  {
    const __bf16* qp = qb + ((size_t)b * T_ + qrow) * H_;
    qop[0] = *(const v16bf*)(qp + 16 * g);
    qop[1] = *(const v16bf*)(qp + 32 + 16 * g);
  }

  const uint64_t kbase  = (uint64_t)(uintptr_t)(kb + (size_t)b * T_ * H_);
  const uint64_t vtbase = (uint64_t)(uintptr_t)(vT + (size_t)b * H_ * T_);
  const int hrow = tid >> 3, hcol = (tid & 7) * 16;

  auto issue = [&](int kv, int bufi) {
    unsigned base = smem_base + bufi * 16384;
    int kgo = kv * H_ * 2;          // K tile: 8KB contiguous at row kv
#pragma unroll
    for (int i = 0; i < 4; ++i)
      async_b128(base + i * 2048 + tid * 16, kgo + i * 2048 + tid * 16, kbase);
    int vgo = kv * 2;               // V^T tile: 64 rows x 128B, row stride T*2 bytes
#pragma unroll
    for (int i = 0; i < 4; ++i)
      async_b128(base + 8192 + (hrow + 16 * i) * 128 + hcol,
                 vgo + (hrow + 16 * i) * (T_ * 2) + hcol, vtbase);
  };

  float m_run = -1.0e30f, l_run = 0.0f;
  v8f o[4] = {};                                   // O: lane = col h, VGPR r = row q
  const int ntiles = qbase / 64 + 1;

  issue(0, 0);
  for (int it = 0; it < ntiles; ++it) {
    wait_async0();
    __syncthreads();
    if (it + 1 < ntiles) issue((it + 1) * 64, (it + 1) & 1);

    const int kv = it * 64;
    const __bf16* Kt = smem + (it & 1) * 8192;     // 64x64 bf16 row-major (keys x dims)
    const __bf16* Vt = Kt + 4096;                  // 64x64 bf16 (h x keys)

    // S^T tiles: 16key x 16q, A = K rows, B = q^T. A ds-loads pipelined 1 ahead.
    v8f st[4];
    auto kptr = [&](int s) {       // s = kt*2 + c
      int kt = s >> 1, c = s & 1;
      return Kt + (16 * kt + l16) * 64 + 32 * c + 16 * g;
    };
    {
      v8bf a0 = *(const v8bf*)kptr(0);
      v8bf a1 = *(const v8bf*)(kptr(0) + 8);
#pragma unroll
      for (int s = 0; s < 8; ++s) {
        v8bf b0, b1;
        if (s < 7) {
          const __bf16* p = kptr(s + 1);
          b0 = *(const v8bf*)p;          // dims 32c+16g .. +7
          b1 = *(const v8bf*)(p + 8);    // dims 32c+16g+8 .. +7
        }
        v16bf A;
        *(v8bf*)&A       = a0;
        *((v8bf*)&A + 1) = a1;
        int kt = s >> 1, c = s & 1;
        v8f z = (c == 0) ? v8f{} : st[kt];
        st[kt] = wmma_bf16(A, qop[c], z);
        a0 = b0; a1 = b1;
      }
    }
    // causal mask (only the diagonal tile)
    if (kv == qbase) {
#pragma unroll
      for (int kt = 0; kt < 4; ++kt)
#pragma unroll
        for (int r = 0; r < 8; ++r)
          if (kv + 16 * kt + r + 8 * g > qrow) st[kt][r] = -1.0e30f;
    }
    // online softmax: each lane owns column q (32 of 64 keys; partner = lane^16)
    float mloc = -1.0e30f;
#pragma unroll
    for (int kt = 0; kt < 4; ++kt)
#pragma unroll
      for (int r = 0; r < 8; ++r) mloc = fmaxf(mloc, st[kt][r]);
    mloc = fmaxf(mloc, __shfl_xor(mloc, 16, 32));
    float mnew  = fmaxf(m_run, mloc);
    float alpha = __expf(m_run - mnew);
    float lloc = 0.0f;
#pragma unroll
    for (int kt = 0; kt < 4; ++kt)
#pragma unroll
      for (int r = 0; r < 8; ++r) {
        float p = __expf(st[kt][r] - mnew);
        st[kt][r] = p;
        lloc += p;
      }
    lloc += __shfl_xor(lloc, 16, 32);
    l_run = l_run * alpha + lloc;
    m_run = mnew;
    // rescale O: O row q = r+8g needs alpha held by lane (r+8g)
#pragma unroll
    for (int r = 0; r < 8; ++r) {
      float ar = __shfl(alpha, r + 8 * g, 32);
#pragma unroll
      for (int ht = 0; ht < 4; ++ht) o[ht][r] *= ar;
    }
    // P (16q x 64key) re-packed in-lane as A-operand chunks
    v16bf P[2];
#pragma unroll
    for (int c = 0; c < 2; ++c)
#pragma unroll
      for (int e = 0; e < 8; ++e) {
        P[c][e]     = (__bf16)st[2 * c][e];
        P[c][8 + e] = (__bf16)st[2 * c + 1][e];
      }
    // O += P * V : B from V^T LDS rows (lane = col h, e<8 -> key 32c+8g+e, e>=8 -> key 32c+16+8g+(e-8))
    auto vptr = [&](int s) {       // s = ht*2 + c
      int ht = s >> 1, c = s & 1;
      return Vt + (16 * ht + l16) * 64 + 32 * c + 8 * g;
    };
    {
      v16bf Bv;
      *(v8bf*)&Bv       = *(const v8bf*)vptr(0);
      *((v8bf*)&Bv + 1) = *(const v8bf*)(vptr(0) + 16);
#pragma unroll
      for (int s = 0; s < 8; ++s) {
        v16bf Bn;
        if (s < 7) {
          *(v8bf*)&Bn       = *(const v8bf*)vptr(s + 1);
          *((v8bf*)&Bn + 1) = *(const v8bf*)(vptr(s + 1) + 16);
        }
        int ht = s >> 1, c = s & 1;
        o[ht] = wmma_bf16(P[c], Bv, o[ht]);
        Bv = Bn;
      }
    }
  }
  // normalize + store: out[b, t, h] f32
  float linv = 1.0f / l_run;
  float* obase = out + (size_t)b * T_ * H_;
#pragma unroll
  for (int r = 0; r < 8; ++r) {
    float li = __shfl(linv, r + 8 * g, 32);
    int trow = qbase + wave * 16 + r + 8 * g;
#pragma unroll
    for (int ht = 0; ht < 4; ++ht)
      obase[(size_t)trow * H_ + 16 * ht + l16] = o[ht][r] * li;
  }
}

// ---------------- Host launcher ------------------------------------------------
extern "C" void kernel_launch(void* const* d_in, const int* in_sizes, int n_in,
                              void* d_out, int out_size, void* d_ws, size_t ws_size,
                              hipStream_t stream) {
  const float* x  = (const float*)d_in[0];
  const float* Wq = (const float*)d_in[1];
  const float* Wk = (const float*)d_in[2];
  const float* Wv = (const float*)d_in[3];
  float* out = (float*)d_out;

  char* ws = (char*)d_ws;
  const size_t qkv_bytes = (size_t)B_ * T_ * H_ * 2;  // 2 MB each
  __bf16* qb = (__bf16*)(ws);
  __bf16* kb = (__bf16*)(ws + qkv_bytes);
  __bf16* vT = (__bf16*)(ws + 2 * qkv_bytes);
  __bf16* Wb = (__bf16*)(ws + 3 * qkv_bytes);        // 384 KB

  wconv_kernel<<<dim3((H_ * C_ + 255) / 256), dim3(256), 0, stream>>>(Wq, Wk, Wv, Wb);
  qkv_kernel<<<dim3((B_ * T_) / 64), dim3(128), 0, stream>>>(x, Wb, qb, kb, vT);
  flash_kernel<<<dim3(T_ / 64, B_), dim3(128), 0, stream>>>(qb, kb, vT, out);
}
